// MambaClassifier_38835094290500
// MI455X (gfx1250) — matrix operand order repaired and checked by reading, not compiled, not run on hardware
//
#include <hip/hip_runtime.h>
#include <hip/hip_bf16.h>

// ---------------------------------------------------------------------------
// Mamba classifier forward for gfx1250 (MI455X).
// GEMMs: v_wmma_f32_16x16x32_bf16. The shared B tile (64 cols x 32 K) is
// staged in LDS, fed by the Tensor Data Mover: each of the 8 waves in a block
// issues a non-divergent tensor_load_to_lds for its own 8-row slice,
// double-buffered, bounded with s_wait_tensorcnt + workgroup barrier.
// K is a template parameter so the whole pipeline is fully unrolled.
// Selective scan is parallel over (batch, channel); only t = L-1 output is
// materialized (the classifier consumes y[:, -1] only).
// ---------------------------------------------------------------------------

#define BATCH    4
#define SEQLEN   2048
#define D_MODEL  512
#define D_INNER  1024
#define D_STATE  16
#define D_CONV   4
#define DT_RANK  32
#define N_CLS    1000
#define MTOT     (BATCH * SEQLEN)        // 8192 rows

typedef __attribute__((ext_vector_type(16))) __bf16        v16bf;
typedef __attribute__((ext_vector_type(8)))  float         v8f;
typedef __attribute__((ext_vector_type(4)))  unsigned int  v4u;
typedef __attribute__((ext_vector_type(8)))  int           v8i;
typedef __attribute__((ext_vector_type(4)))  int           v4i;

union Frag32 { uint4 u[2]; v16bf v; };   // 32 bytes: one 16-bf16 fragment half

#if __has_builtin(__builtin_amdgcn_tensor_load_to_lds) && \
    __has_builtin(__builtin_amdgcn_s_wait_tensorcnt)
#define USE_TDM 1
#endif

// LDS B-tile row: 32 bf16 payload + 8 bf16 pad = 80B (20 dwords) per row.
// TDM pad descriptor: pad_interval code 3 (every 16 dwords) + pad_amount
// code 3 (4 dwords) reproduces exactly this layout.
#define BROW 40

__device__ __forceinline__ unsigned short f32_to_bf16(float f) {
    unsigned int u = __float_as_uint(f);
    unsigned int lsb = (u >> 16) & 1u;
    u += 0x7fffu + lsb;                  // round-to-nearest-even
    return (unsigned short)(u >> 16);
}

__device__ __forceinline__ float sigmoidf_(float x) {
    return 1.0f / (1.0f + __expf(-x));
}

__device__ __forceinline__ unsigned int rfl_u32(unsigned int x) {
    return (unsigned int)__builtin_amdgcn_readfirstlane((int)x);
}

// ---------------------------------------------------------------------------
// Block GEMM: 256 threads (8 waves), block tile 128M x 64N, K steps of 32.
// Wave w owns rows m0+16w..+15; all 8 waves share the LDS-staged B tile.
// lda == ldw == K (all operands are row-major [rows, K]).
// EPI: 0 = plain store, 1 = store + dt bf16 split, 2 = bias + softplus.
// ---------------------------------------------------------------------------
template<int EPI, int K>
__global__ __launch_bounds__(256)
void k_gemm_lds(const unsigned short* __restrict__ A,
                const unsigned short* __restrict__ W,
                float* __restrict__ OUT, int ldo,
                const float* __restrict__ bias,
                unsigned short* __restrict__ OUT2)
{
    __shared__ unsigned short Bs[2][64][BROW];

    constexpr int NSTEPS = K / 32;
    const int tid  = threadIdx.x;
    const int lane = tid & 31;
    const int wid  = tid >> 5;           // 0..7
    const int half = lane >> 4;          // 0 | 1
    const int l16  = lane & 15;
    const int m0   = blockIdx.x * 128 + wid * 16;
    const int n0   = blockIdx.y * 64;

    v8f acc[4];
#pragma unroll
    for (int j = 0; j < 4; ++j) acc[j] = 0.0f;

    const unsigned short* arow = A + (long)(m0 + l16) * K;

#ifdef USE_TDM
    // ---- TDM: every wave copies its own 8-row slice of the 64x32 B tile.
    // Non-divergent (whole block executes); per-wave descriptor words are
    // pinned to SGPRs with readfirstlane. Per-wave TENSORcnt bounds its own
    // in-order copies; the workgroup barrier publishes all slices.
    auto issue_stage = [&](int s, int buf) {
        unsigned long long ga = (unsigned long long)(size_t)(const void*)
            (W + (long)(n0 + wid * 8) * K + s * 32);
        unsigned int galo = rfl_u32((unsigned int)ga);
        unsigned int gahi = rfl_u32((unsigned int)(ga >> 32));
        unsigned int ldsb = rfl_u32((unsigned int)(size_t)(void*)&Bs[buf][wid * 8][0]);
        v4u g0;
        g0[0] = 1u;                                  // count=1, user mode
        g0[1] = ldsb;                                // lds_addr
        g0[2] = galo;                                // global_addr[31:0]
        g0[3] = (gahi & 0x01ffffffu) | (2u << 30);   // addr[56:32] | type=2
        v8i g1;
        g1[0] = (int)((1u << 16) |                   // data_size = 2 bytes
                      (1u << 20) |                   // pad_enable
                      (3u << 22) |                   // pad_interval: 16 dwords
                      (3u << 25));                   // pad_amount: 4 dwords
        g1[1] = (int)(((unsigned)K & 0xffffu) << 16);          // tensor_dim0 lo
        g1[2] = (int)((((unsigned)K >> 16) & 0xffffu) |        // tensor_dim0 hi
                      (8u << 16));                             // tensor_dim1 = 8
        g1[3] = (int)(32u << 16);                              // tile_dim0 = 32
        g1[4] = 8;                                             // tile_dim1 = 8
        g1[5] = K;                                             // dim0_stride lo
        g1[6] = 0;
        g1[7] = 0;
        v4i gz = {0, 0, 0, 0};
#if __clang_major__ >= 23
        v8i gz8 = {0, 0, 0, 0, 0, 0, 0, 0};
        __builtin_amdgcn_tensor_load_to_lds(g0, g1, gz, gz, gz8, 0);
#else
        __builtin_amdgcn_tensor_load_to_lds(g0, g1, gz, gz, 0);
#endif
    };

    issue_stage(0, 0);
#pragma unroll
    for (int s = 0; s < NSTEPS; ++s) {
        const int buf = s & 1;
        if (s + 1 < NSTEPS) {            // folds at compile time (full unroll)
            issue_stage(s + 1, buf ^ 1);
            __builtin_amdgcn_s_wait_tensorcnt(1);   // stage s done (per wave)
        } else {
            __builtin_amdgcn_s_wait_tensorcnt(0);
        }
        __syncthreads();                 // all 8 slices of stage s visible

        Frag32 a;
        const unsigned short* ap = arow + s * 32 + half * 8;
        a.u[0] = *(const uint4*)(ap);
        a.u[1] = *(const uint4*)(ap + 16);
        if (s + 1 < NSTEPS) __builtin_prefetch(ap + 32, 0, 1);
#pragma unroll
        for (int j = 0; j < 4; ++j) {
            Frag32 bfr;
            const unsigned short* bp = &Bs[buf][j * 16 + l16][half * 8];
            bfr.u[0] = *(const uint4*)(bp);
            bfr.u[1] = *(const uint4*)(bp + 16);
            acc[j] = __builtin_amdgcn_wmma_f32_16x16x32_bf16(
                false, a.v, false, bfr.v, (short)0, acc[j], false, false);
        }
        __syncthreads();                 // readers done before buf is reused
    }
#else
    // ---- Fallback: synchronous global->reg->LDS double buffer.
    // 256 threads stage 64 rows x 32 bf16: thread t copies one uint4.
    uint4 sreg;
    auto load_stage = [&](int s) {
        const unsigned short* src =
            W + (long)(n0 + (tid >> 2)) * K + s * 32 + (tid & 3) * 8;
        sreg = *(const uint4*)(src);
    };
    auto store_stage = [&](int buf) {
        *(uint4*)&Bs[buf][tid >> 2][(tid & 3) * 8] = sreg;
    };
    load_stage(0);
    store_stage(0);
    __syncthreads();
#pragma unroll
    for (int s = 0; s < NSTEPS; ++s) {
        const int buf = s & 1;
        if (s + 1 < NSTEPS) load_stage(s + 1);

        Frag32 a;
        const unsigned short* ap = arow + s * 32 + half * 8;
        a.u[0] = *(const uint4*)(ap);
        a.u[1] = *(const uint4*)(ap + 16);
        if (s + 1 < NSTEPS) __builtin_prefetch(ap + 32, 0, 1);
#pragma unroll
        for (int j = 0; j < 4; ++j) {
            Frag32 bfr;
            const unsigned short* bp = &Bs[buf][j * 16 + l16][half * 8];
            bfr.u[0] = *(const uint4*)(bp);
            bfr.u[1] = *(const uint4*)(bp + 16);
            acc[j] = __builtin_amdgcn_wmma_f32_16x16x32_bf16(
                false, a.v, false, bfr.v, (short)0, acc[j], false, false);
        }
        __syncthreads();
        if (s + 1 < NSTEPS) store_stage(buf ^ 1);
        __syncthreads();
    }
#endif

    // ---- epilogue: C/D layout -> row m0 + half*8 + r, col n0 + j*16 + l16
#pragma unroll
    for (int j = 0; j < 4; ++j)
#pragma unroll
        for (int r = 0; r < 8; ++r) {
            int row = m0 + half * 8 + r;
            int col = n0 + j * 16 + l16;
            float v = acc[j][r];
            if (EPI == 2) {
                v += bias[col];
                v = (v > 20.0f) ? v : log1pf(__expf(v));   // softplus
            }
            OUT[(long)row * ldo + col] = v;
            if (EPI == 1 && col < DT_RANK)
                OUT2[(long)row * DT_RANK + col] = f32_to_bf16(v);
        }
}

// -------------------------------- other kernels ----------------------------

__global__ void k_cast_bf16(const float* __restrict__ src,
                            unsigned short* __restrict__ dst, int n) {
    int i = blockIdx.x * blockDim.x + threadIdx.x;
    if (i < n) dst[i] = f32_to_bf16(src[i]);
}

// x = emb[tokens] cast to bf16, row-major [MTOT, D_MODEL]
__global__ void k_embed(const int* __restrict__ tokens,
                        const float* __restrict__ emb,
                        unsigned short* __restrict__ Xb) {
    int m = blockIdx.x;
    int t = tokens[m];
    const float* src = emb + (long)t * D_MODEL;
    for (int k = threadIdx.x; k < D_MODEL; k += blockDim.x)
        Xb[(long)m * D_MODEL + k] = f32_to_bf16(src[k]);
}

// causal depthwise conv (k=4) + bias + SiLU over xin = xz[:, :1024]
__global__ void k_conv(const float* __restrict__ XZ,
                       const float* __restrict__ cw,
                       const float* __restrict__ cb,
                       float* __restrict__ XC,
                       unsigned short* __restrict__ XCb) {
    int m = blockIdx.x;                  // b*L + l
    int l = m & (SEQLEN - 1);
    for (int d = threadIdx.x; d < D_INNER; d += blockDim.x) {
        float acc = cb[d];
#pragma unroll
        for (int k = 0; k < D_CONV; ++k) {
            int lk = l + k - (D_CONV - 1);
            if (lk >= 0)
                acc += XZ[(long)(m + k - (D_CONV - 1)) * (2 * D_INNER) + d] *
                       cw[d * D_CONV + k];
        }
        float s = acc * sigmoidf_(acc);  // SiLU
        XC [(long)m * D_INNER + d] = s;
        XCb[(long)m * D_INNER + d] = f32_to_bf16(s);
    }
}

// selective scan: one thread per (b, d) channel; 16-state recurrence in regs.
// Only t = L-1 emits output: y = (h . C) + xc*D, gated by silu(z).
__global__ __launch_bounds__(256)
void k_scan(const float* __restrict__ DTF, const float* __restrict__ XDBL,
            const float* __restrict__ XC,  const float* __restrict__ XZ,
            const float* __restrict__ Alog, const float* __restrict__ Dp,
            float* __restrict__ Ylast) {
    const int b = blockIdx.y;
    const int d = blockIdx.x * blockDim.x + threadIdx.x;   // 0..1023
    float a[D_STATE], h[D_STATE];
#pragma unroll
    for (int s = 0; s < D_STATE; ++s) {
        a[s] = -__expf(Alog[d * D_STATE + s]);
        h[s] = 0.0f;
    }
    const long base = (long)b * SEQLEN;
    for (int t = 0; t < SEQLEN; ++t) {
        const long m = base + t;
        float dt = DTF[m * D_INNER + d];
        float x  = XC [m * D_INNER + d];
        float dx = dt * x;
        float Bv[D_STATE];
#pragma unroll
        for (int i = 0; i < 4; ++i) {
            float4 q = *(const float4*)(XDBL + m * 64 + DT_RANK + 4 * i);
            Bv[4 * i + 0] = q.x; Bv[4 * i + 1] = q.y;
            Bv[4 * i + 2] = q.z; Bv[4 * i + 3] = q.w;
        }
#pragma unroll
        for (int s = 0; s < D_STATE; ++s)
            h[s] = __expf(dt * a[s]) * h[s] + dx * Bv[s];
    }
    const long mL = base + (SEQLEN - 1);
    float y = 0.0f;
#pragma unroll
    for (int i = 0; i < 4; ++i) {
        float4 q = *(const float4*)(XDBL + mL * 64 + DT_RANK + D_STATE + 4 * i);
        y += h[4*i+0] * q.x + h[4*i+1] * q.y + h[4*i+2] * q.z + h[4*i+3] * q.w;
    }
    y += XC[mL * D_INNER + d] * Dp[d];
    float z = XZ[mL * (2 * D_INNER) + D_INNER + d];
    y *= z * sigmoidf_(z);
    Ylast[b * D_INNER + d] = y;
}

// last[4, 512] = y_last @ out_proj_w^T   (tiny: scalar fp32)
__global__ void k_outproj(const float* __restrict__ Y,
                          const float* __restrict__ Wo,
                          float* __restrict__ Last) {
    int b = blockIdx.y;
    int e = blockIdx.x * blockDim.x + threadIdx.x;   // 0..511
    float acc = 0.0f;
    for (int d = 0; d < D_INNER; ++d)
        acc += Y[b * D_INNER + d] * Wo[(long)e * D_INNER + d];
    Last[b * D_MODEL + e] = acc;
}

// out[4, 1000] = last @ cls_w^T + cls_b
__global__ void k_cls(const float* __restrict__ Last,
                      const float* __restrict__ Wc,
                      const float* __restrict__ bc,
                      float* __restrict__ Out) {
    int b = blockIdx.y;
    int e = blockIdx.x * blockDim.x + threadIdx.x;
    if (e < N_CLS) {
        float acc = bc[e];
        for (int k = 0; k < D_MODEL; ++k)
            acc += Last[b * D_MODEL + k] * Wc[(long)e * D_MODEL + k];
        Out[b * N_CLS + e] = acc;
    }
}

// ------------------------------ launcher -----------------------------------

extern "C" void kernel_launch(void* const* d_in, const int* in_sizes, int n_in,
                              void* d_out, int out_size, void* d_ws, size_t ws_size,
                              hipStream_t stream) {
    const int*   tokens    = (const int*)  d_in[0];
    const float* emb       = (const float*)d_in[1];
    const float* in_proj_w = (const float*)d_in[2];
    const float* conv_w    = (const float*)d_in[3];
    const float* conv_b    = (const float*)d_in[4];
    const float* x_proj_w  = (const float*)d_in[5];
    const float* dt_proj_w = (const float*)d_in[6];
    const float* dt_proj_b = (const float*)d_in[7];
    const float* A_log     = (const float*)d_in[8];
    const float* Dparm     = (const float*)d_in[9];
    const float* out_proj_w= (const float*)d_in[10];
    const float* cls_w     = (const float*)d_in[11];
    const float* cls_b     = (const float*)d_in[12];
    float* out = (float*)d_out;

    // scratch layout (256B aligned slabs)
    char* ws = (char*)d_ws;
    size_t off = 0;
    auto alloc = [&](size_t bytes) -> char* {
        char* p = ws + off;
        off = (off + bytes + 255) & ~(size_t)255;
        return p;
    };
    unsigned short* Xb   = (unsigned short*)alloc((size_t)MTOT * D_MODEL * 2);
    unsigned short* Wib  = (unsigned short*)alloc((size_t)2 * D_INNER * D_MODEL * 2);
    unsigned short* Wxb  = (unsigned short*)alloc((size_t)64 * D_INNER * 2);
    unsigned short* Wdb  = (unsigned short*)alloc((size_t)D_INNER * DT_RANK * 2);
    float*          XZ   = (float*)alloc((size_t)MTOT * 2 * D_INNER * 4);
    float*          XC   = (float*)alloc((size_t)MTOT * D_INNER * 4);
    unsigned short* XCb  = (unsigned short*)alloc((size_t)MTOT * D_INNER * 2);
    float*          XDBL = (float*)alloc((size_t)MTOT * 64 * 4);
    unsigned short* DTb  = (unsigned short*)alloc((size_t)MTOT * DT_RANK * 2);
    float*          DTF  = (float*)alloc((size_t)MTOT * D_INNER * 4);
    float*          Yl   = (float*)alloc((size_t)BATCH * D_INNER * 4);
    float*          Last = (float*)alloc((size_t)BATCH * D_MODEL * 4);
    (void)ws_size; (void)n_in; (void)in_sizes; (void)out_size;

    // weight casts to bf16
    {
        int n1 = 2 * D_INNER * D_MODEL;
        k_cast_bf16<<<(n1 + 255) / 256, 256, 0, stream>>>(in_proj_w, Wib, n1);
        int n2 = 64 * D_INNER;
        k_cast_bf16<<<(n2 + 255) / 256, 256, 0, stream>>>(x_proj_w, Wxb, n2);
        int n3 = D_INNER * DT_RANK;
        k_cast_bf16<<<(n3 + 255) / 256, 256, 0, stream>>>(dt_proj_w, Wdb, n3);
    }
    // embedding gather + bf16 cast
    k_embed<<<MTOT, 256, 0, stream>>>(tokens, emb, Xb);

    // in_proj GEMM: [8192,512] x [2048,512]^T -> [8192,2048]
    k_gemm_lds<0, D_MODEL><<<dim3(MTOT / 128, (2 * D_INNER) / 64), 256, 0, stream>>>(
        Xb, Wib, XZ, 2 * D_INNER, nullptr, nullptr);

    // conv + SiLU
    k_conv<<<MTOT, 256, 0, stream>>>(XZ, conv_w, conv_b, XC, XCb);

    // x_proj GEMM: [8192,1024] x [64,1024]^T -> [8192,64]  (+ dt bf16 split)
    k_gemm_lds<1, D_INNER><<<dim3(MTOT / 128, 1), 256, 0, stream>>>(
        XCb, Wxb, XDBL, 64, nullptr, DTb);

    // dt_proj GEMM + softplus: [8192,32] x [1024,32]^T -> [8192,1024]
    k_gemm_lds<2, DT_RANK><<<dim3(MTOT / 128, D_INNER / 64), 256, 0, stream>>>(
        DTb, Wdb, DTF, D_INNER, dt_proj_b, nullptr);

    // selective scan (parallel over batch x channel, sequential over time)
    k_scan<<<dim3(D_INNER / 256, BATCH), 256, 0, stream>>>(DTF, XDBL, XC, XZ, A_log, Dparm, Yl);

    // out_proj on last token, then classifier head
    k_outproj<<<dim3(D_MODEL / 256, BATCH), 256, 0, stream>>>(Yl, out_proj_w, Last);
    k_cls<<<dim3((N_CLS + 255) / 256, BATCH), 256, 0, stream>>>(Last, cls_w, cls_b, out);
}
